// DotProductAttention_48421461295297
// MI455X (gfx1250) — compile-verified, matrix operand order checked by reading, not run on hardware
//
#include <hip/hip_runtime.h>

typedef _Float16 v16h __attribute__((ext_vector_type(16)));
typedef _Float16 v8h  __attribute__((ext_vector_type(8)));
typedef float    v8f  __attribute__((ext_vector_type(8)));

constexpr int BB = 16;
constexpr int SS = 2048;
constexpr int DD = 128;
constexpr int KVT = 32;              // kv rows per tile
constexpr int NT  = SS / KVT;        // 64 kv tiles
constexpr int WAVES = 4;             // wave32s per workgroup
constexpr int TILE_H = KVT * DD;     // f16 elements per staged tile (4096 = 8KB)

// gfx1250 async global->LDS copy path (ASYNCcnt) if the toolchain exposes it.
#if defined(__has_builtin)
#if __has_builtin(__builtin_amdgcn_global_load_async_to_lds_b128)
#define USE_ASYNC_LDS 1
#endif
#endif

#ifdef USE_ASYNC_LDS
// Param 0: gcc-style int4 pointer in AS(1) (global); param 1: LDS-side pointer.
typedef int v4i __attribute__((vector_size(16)));
typedef __attribute__((address_space(1))) v4i* gasp;   // global (AS1)
typedef __attribute__((address_space(3))) v4i* lasp;   // LDS (AS3)
__device__ __forceinline__ gasp to_global(const void* p) {
    return (gasp)(unsigned long long)p;                // AS1 == generic numerically
}
__device__ __forceinline__ lasp to_lds(void* p) {
    return (lasp)(unsigned)(unsigned long long)p;      // LDS offset = low 32 bits
}
__device__ __forceinline__ void wait_async0() {
#if __has_builtin(__builtin_amdgcn_s_wait_asynccnt)
    __builtin_amdgcn_s_wait_asynccnt(0);
#else
    asm volatile("s_wait_asynccnt 0x0" ::: "memory");
#endif
}
#endif

// Raw v_exp_f32: args here are always <= 0, and sub-(-126) inputs should flush
// to 0 probability anyway, so skip ocml's denormal-range fixup entirely.
__device__ __forceinline__ float fast_exp2(float x) {
#if defined(__has_builtin) && __has_builtin(__builtin_amdgcn_exp2f)
    return __builtin_amdgcn_exp2f(x);
#else
    return exp2f(x);
#endif
}
__device__ __forceinline__ float fast_rcp(float x) {
#if defined(__has_builtin) && __has_builtin(__builtin_amdgcn_rcpf)
    return __builtin_amdgcn_rcpf(x);
#else
    return 1.0f / x;
#endif
}

// Build a v16h WMMA operand from two contiguous 16B LDS chunks (2x ds_load_b128).
__device__ __forceinline__ v16h ld_op16(const _Float16* p) {
    v8h lo = *(const v8h*)p;
    v8h hh = *(const v8h*)(p + 16);
    v16h r;
#pragma unroll
    for (int i = 0; i < 8; ++i) { r[i] = lo[i]; r[i + 8] = hh[i]; }
    return r;
}

// ---------------- pre-pass: K -> f16 flat, V -> f16 transposed per 32-row tile ----------------
__global__ __launch_bounds__(128, 1)
void fa_prep(const float* __restrict__ K, const float* __restrict__ V,
             _Float16* __restrict__ Kh, _Float16* __restrict__ Vt)
{
    __shared__ __align__(16) _Float16 T[DD * KVT];   // [d][k]
    const int tid  = threadIdx.x;
    const int b    = blockIdx.x >> 6;
    const int t    = blockIdx.x & 63;
    const int krow = tid >> 2;
    const int cgrp = (tid & 3) * 32;
    const size_t tb = ((size_t)b * SS + (size_t)t * KVT) * DD;
    const float* kp = K + tb + (size_t)krow * DD + cgrp;
    const float* vp = V + tb + (size_t)krow * DD + cgrp;
    _Float16*    kq = Kh + tb + (size_t)krow * DD + cgrp;
#pragma unroll
    for (int j = 0; j < 4; ++j) {
        float4 a = *(const float4*)(kp + 8 * j);
        float4 c = *(const float4*)(kp + 8 * j + 4);
        v8h h;
        h[0]=(_Float16)a.x; h[1]=(_Float16)a.y; h[2]=(_Float16)a.z; h[3]=(_Float16)a.w;
        h[4]=(_Float16)c.x; h[5]=(_Float16)c.y; h[6]=(_Float16)c.z; h[7]=(_Float16)c.w;
        *(v8h*)(kq + 8 * j) = h;
        float4 e = *(const float4*)(vp + 8 * j);
        float4 f = *(const float4*)(vp + 8 * j + 4);
        const int d0 = cgrp + 8 * j;
        T[(d0+0)*KVT + krow] = (_Float16)e.x;
        T[(d0+1)*KVT + krow] = (_Float16)e.y;
        T[(d0+2)*KVT + krow] = (_Float16)e.z;
        T[(d0+3)*KVT + krow] = (_Float16)e.w;
        T[(d0+4)*KVT + krow] = (_Float16)f.x;
        T[(d0+5)*KVT + krow] = (_Float16)f.y;
        T[(d0+6)*KVT + krow] = (_Float16)f.z;
        T[(d0+7)*KVT + krow] = (_Float16)f.w;
    }
    __syncthreads();
    _Float16* vo = Vt + tb + (size_t)tid * KVT;
    const _Float16* ts = &T[tid * KVT];
#pragma unroll
    for (int j = 0; j < 4; ++j)
        *(v8h*)(vo + 8 * j) = *(const v8h*)(ts + 8 * j);
}

// ---------------- main FlashAttention kernel (transpose-free, double-buffered) ----------------
__global__ __launch_bounds__(128, 1)
void DotProductAttention_48421461295297_kernel(const float* __restrict__ Q,
                                               const _Float16* __restrict__ Kh,
                                               const _Float16* __restrict__ Vt,
                                               float* __restrict__ O)
{
    __shared__ __align__(16) _Float16 Kst[2][TILE_H];   // [k=32][d=128]
    __shared__ __align__(16) _Float16 Vst[2][TILE_H];   // [d=128][k=32]

    const int tid  = threadIdx.x;
    const int wave = tid >> 5;
    const int lane = tid & 31;
    const int ln   = lane & 15;
    const int koff = (lane >> 4) * 8;   // K-range offset in operand layout
    const int mb   = koff;              // row-base in C/D layout

    const int batch = blockIdx.x >> 5;
    const int qblk  = blockIdx.x & 31;
    const int q0    = qblk * (WAVES * 16) + wave * 16;
    const size_t bbase = (size_t)batch * SS * DD;

    const float SL2E = 0.08838834764831845f * 1.4426950408889634f;  // 1/sqrt(128)*log2(e)

    // Q tile -> f16 registers with the softmax scale folded in (log2 domain).
    v16h qa[4];
    {
        const float* qp = Q + bbase + (size_t)(q0 + ln) * DD;
#pragma unroll
        for (int dc = 0; dc < 4; ++dc) {
            const int off = dc * 32 + koff;
            float4 f0 = *(const float4*)(qp + off);
            float4 f1 = *(const float4*)(qp + off + 4);
            float4 f2 = *(const float4*)(qp + off + 16);
            float4 f3 = *(const float4*)(qp + off + 20);
            v16h a;
            a[0]=(_Float16)(f0.x*SL2E);  a[1]=(_Float16)(f0.y*SL2E);
            a[2]=(_Float16)(f0.z*SL2E);  a[3]=(_Float16)(f0.w*SL2E);
            a[4]=(_Float16)(f1.x*SL2E);  a[5]=(_Float16)(f1.y*SL2E);
            a[6]=(_Float16)(f1.z*SL2E);  a[7]=(_Float16)(f1.w*SL2E);
            a[8]=(_Float16)(f2.x*SL2E);  a[9]=(_Float16)(f2.y*SL2E);
            a[10]=(_Float16)(f2.z*SL2E); a[11]=(_Float16)(f2.w*SL2E);
            a[12]=(_Float16)(f3.x*SL2E); a[13]=(_Float16)(f3.y*SL2E);
            a[14]=(_Float16)(f3.z*SL2E); a[15]=(_Float16)(f3.w*SL2E);
            qa[dc] = a;
        }
    }

    v8f ot[8] = {};                       // O^T accumulators: rows d, cols q (fp32)
    float mrun = -__builtin_inff();
    float lrun = 0.0f;

    // staging: each thread owns 64B of the K tile and 64B of the V tile
    const _Float16* kbase = Kh + bbase + (size_t)tid * 32;
    const _Float16* vbase = Vt + bbase + (size_t)tid * 32;

#ifdef USE_ASYNC_LDS
    {   // stage tile 0: direct global->LDS, no VGPR traffic
        gasp kg = to_global(kbase);
        gasp vg = to_global(vbase);
        lasp kl = to_lds(&Kst[0][tid * 32]);
        lasp vl = to_lds(&Vst[0][tid * 32]);
#pragma unroll
        for (int j = 0; j < 4; ++j) {
            __builtin_amdgcn_global_load_async_to_lds_b128(kg + j, kl + j, 0, 0);
            __builtin_amdgcn_global_load_async_to_lds_b128(vg + j, vl + j, 0, 0);
        }
        wait_async0();
    }
#else
    v8h kr[4], vr[4];
#pragma unroll
    for (int j = 0; j < 4; ++j) {
        kr[j] = *(const v8h*)(kbase + 8 * j);
        vr[j] = *(const v8h*)(vbase + 8 * j);
    }
#pragma unroll
    for (int j = 0; j < 4; ++j) {
        *(v8h*)(&Kst[0][tid * 32 + 8 * j]) = kr[j];
        *(v8h*)(&Vst[0][tid * 32 + 8 * j]) = vr[j];
    }
#endif
    __syncthreads();

#pragma unroll 1
    for (int t = 0; t < NT; ++t) {
        const int cur = t & 1;

        if (t + 1 < NT) {                 // start next tile's copy before compute
#ifdef USE_ASYNC_LDS
            gasp kg = to_global(kbase + (size_t)(t + 1) * TILE_H);
            gasp vg = to_global(vbase + (size_t)(t + 1) * TILE_H);
            lasp kl = to_lds(&Kst[cur ^ 1][tid * 32]);
            lasp vl = to_lds(&Vst[cur ^ 1][tid * 32]);
#pragma unroll
            for (int j = 0; j < 4; ++j) {
                __builtin_amdgcn_global_load_async_to_lds_b128(kg + j, kl + j, 0, 0);
                __builtin_amdgcn_global_load_async_to_lds_b128(vg + j, vl + j, 0, 0);
            }
#else
            const _Float16* kn = kbase + (size_t)(t + 1) * TILE_H;
            const _Float16* vn = vbase + (size_t)(t + 1) * TILE_H;
#pragma unroll
            for (int j = 0; j < 4; ++j) {
                kr[j] = *(const v8h*)(kn + 8 * j);
                vr[j] = *(const v8h*)(vn + 8 * j);
            }
#endif
            if (t + 2 < NT) {             // gfx1250 global_prefetch of tile t+2
                __builtin_prefetch(kbase + (size_t)(t + 2) * TILE_H, 0, 1);
                __builtin_prefetch(vbase + (size_t)(t + 2) * TILE_H, 0, 1);
            }
        }

        // ---- S^T = K_tile * Q^T ----
        v16h kb[8];
#pragma unroll
        for (int dc = 0; dc < 4; ++dc) {
            kb[dc]     = ld_op16(&Kst[cur][(0  + ln) * DD + dc * 32 + koff]);
            kb[4 + dc] = ld_op16(&Kst[cur][(16 + ln) * DD + dc * 32 + koff]);
        }
        v8f ct0 = {}, ct1 = {};
#pragma unroll
        for (int dc = 0; dc < 4; ++dc) {
            ct0 = __builtin_amdgcn_wmma_f32_16x16x32_f16(false, kb[dc],     false, qa[dc],
                                                         (short)0, ct0, false, false);
            ct1 = __builtin_amdgcn_wmma_f32_16x16x32_f16(false, kb[4 + dc], false, qa[dc],
                                                         (short)0, ct1, false, false);
        }

        // ---- online softmax (log2 domain; scores already scaled via Q) ----
        float vmax = ct0[0];
#pragma unroll
        for (int r = 1; r < 8; ++r) vmax = fmaxf(vmax, ct0[r]);
#pragma unroll
        for (int r = 0; r < 8; ++r) vmax = fmaxf(vmax, ct1[r]);
        vmax = fmaxf(vmax, __shfl_xor(vmax, 16, 32));
        const float mnew = fmaxf(mrun, vmax);
        const float corr = fast_exp2(mrun - mnew);
        mrun = mnew;

        v16h pa;                          // exponentiated scores == P^T B-operand layout
        float rs = 0.0f;
#pragma unroll
        for (int r = 0; r < 8; ++r) { float p = fast_exp2(ct0[r] - mnew); rs += p; pa[r]     = (_Float16)p; }
#pragma unroll
        for (int r = 0; r < 8; ++r) { float p = fast_exp2(ct1[r] - mnew); rs += p; pa[8 + r] = (_Float16)p; }
        rs += __shfl_xor(rs, 16, 32);
        lrun = lrun * corr + rs;

#pragma unroll
        for (int db = 0; db < 8; ++db)    // v8f * scalar -> v_pk_mul_f32
            ot[db] = ot[db] * corr;

        // ---- O^T += V^T * P^T ----
        v16h vb[8];
#pragma unroll
        for (int db = 0; db < 8; ++db)
            vb[db] = ld_op16(&Vst[cur][(db * 16 + ln) * KVT + koff]);
#pragma unroll
        for (int db = 0; db < 8; ++db)
            ot[db] = __builtin_amdgcn_wmma_f32_16x16x32_f16(false, vb[db], false, pa,
                                                            (short)0, ot[db], false, false);

        if (t + 1 < NT) {
#ifdef USE_ASYNC_LDS
            wait_async0();                // drain our async writes before publishing
#else
            _Float16* kd = &Kst[cur ^ 1][tid * 32];
            _Float16* vd = &Vst[cur ^ 1][tid * 32];
#pragma unroll
            for (int j = 0; j < 4; ++j) {
                *(v8h*)(kd + 8 * j) = kr[j];
                *(v8h*)(vd + 8 * j) = vr[j];
            }
#endif
        }
        __syncthreads();
    }

    // ---- normalize + store: lane writes q-row ln, d = db*16 + mb + 0..7 ----
    const float inv = fast_rcp(lrun);
    float* orow = O + bbase + (size_t)(q0 + ln) * DD + mb;
#pragma unroll
    for (int db = 0; db < 8; ++db) {
        float4 lo4, hi4;
        lo4.x = ot[db][0] * inv; lo4.y = ot[db][1] * inv;
        lo4.z = ot[db][2] * inv; lo4.w = ot[db][3] * inv;
        hi4.x = ot[db][4] * inv; hi4.y = ot[db][5] * inv;
        hi4.z = ot[db][6] * inv; hi4.w = ot[db][7] * inv;
        *(float4*)(orow + db * 16)     = lo4;
        *(float4*)(orow + db * 16 + 4) = hi4;
    }
}

extern "C" void kernel_launch(void* const* d_in, const int* in_sizes, int n_in,
                              void* d_out, int out_size, void* d_ws, size_t ws_size,
                              hipStream_t stream) {
    (void)in_sizes; (void)n_in; (void)out_size; (void)ws_size;
    const float* Q = (const float*)d_in[0];
    const float* K = (const float*)d_in[1];
    const float* V = (const float*)d_in[2];
    float* O = (float*)d_out;
    // workspace: Kh (8.39 MB) | Vt (8.39 MB)  -> 16.78 MB of d_ws
    _Float16* Kh = (_Float16*)d_ws;
    _Float16* Vt = Kh + (size_t)BB * SS * DD;
    hipLaunchKernelGGL(fa_prep, dim3(BB * NT), dim3(128), 0, stream, K, V, Kh, Vt);
    hipLaunchKernelGGL(DotProductAttention_48421461295297_kernel,
                       dim3(BB * (SS / (WAVES * 16))), dim3(128), 0, stream, Q, Kh, Vt, O);
}
